// ULSTM_73186242724132
// MI455X (gfx1250) — compile-verified, matrix operand order checked by reading, not run on hardware
//
#include <hip/hip_runtime.h>
#include <hip/hip_bf16.h>
#include <stdint.h>

// ---------------------------------------------------------------------------
// ULSTM for MI455X (gfx1250, wave32)
//   Part 1: xproj GEMM (8192x5120x1024): TDM tensor_load_to_lds tile staging
//           (double-buffered, TENSORcnt-pipelined) -> v_wmma_f32_16x16x32_bf16
//   Part 2: serial scan (8192 steps): persistent grid, fp32 VALU matvecs,
//           h/a vectors staged to LDS via global_load_async_to_lds_b128
// ---------------------------------------------------------------------------

#define IN_DIM   1024
#define MEM_DIM  1024
#define T_STEPS  8192
#define NGATE5   (5 * MEM_DIM)   // 5120
#define NGATE4   (4 * MEM_DIM)   // 4096

typedef __attribute__((ext_vector_type(16))) __bf16 v16bf;
typedef __attribute__((ext_vector_type(8)))  __bf16 v8bf;
typedef __attribute__((ext_vector_type(8)))  float  v8f;
typedef __attribute__((ext_vector_type(4)))  unsigned int u32x4;
typedef __attribute__((ext_vector_type(8)))  unsigned int u32x8;

// Low 32 bits of a generic pointer into the LDS aperture == wave-relative
// LDS byte offset (flat addressing truncates addr[31:0] for the LDS space).
__device__ __forceinline__ unsigned lds_off(const void* p) {
  return (unsigned)(unsigned long long)p;
}

// ---------------------------------------------------------------------------
// fp32 -> bf16 conversion (tiny vs. GEMM cost)
// ---------------------------------------------------------------------------
__global__ void f32_to_bf16_kernel(const float* __restrict__ in,
                                   unsigned short* __restrict__ out, int n) {
  int i = blockIdx.x * blockDim.x + threadIdx.x;
  int stride = gridDim.x * blockDim.x;
  for (; i < n; i += stride) {
    __hip_bfloat16 b = __float2bfloat16(in[i]);
    out[i] = *reinterpret_cast<unsigned short*>(&b);
  }
}

// ---------------------------------------------------------------------------
// TDM: issue a 2D tile load (2-byte elements) global -> LDS.
// D# per CDNA5 ISA. Groups 2/3 omitted (2D).
// Hardware pad: after every pad_interval DWORDs of a row, skip pad_amount
// DWORDs in LDS -> implements the +16B row padding for bank-conflict-free
// ds_load_b128 fragment reads.
//   tile_w elems/row (row = tile_w*2 bytes), tile_h rows,
//   row_stride in elements between consecutive tensor rows.
// Issued once per wave (EXEC ignored by TDM); call from one wave only.
// ---------------------------------------------------------------------------
__device__ __forceinline__ void tdm_load_tile_bf16(
    unsigned lds_byte_addr, const unsigned short* global_tile,
    unsigned tile_w, unsigned tile_h, unsigned row_stride,
    unsigned pad_interval_code, unsigned pad_amount_code) {
  unsigned long long ga = (unsigned long long)global_tile;
  u32x4 g0;
  g0.x = 1u;                                    // count=1, user-mode D#
  g0.y = lds_byte_addr;                         // bits 63:32 lds_addr
  g0.z = (unsigned)(ga & 0xFFFFFFFFu);          // bits 95:64 global_addr lo
  g0.w = (unsigned)((ga >> 32) & 0x01FFFFFFu)   // bits 120:96 global_addr hi
         | (2u << 30);                          // bits 127:126 type=2 (image)
  u32x8 g1;
  g1.s0 = (1u << 16)                            // data_size=1 (2 bytes)
        | (1u << 20)                            // pad_enable
        | (pad_interval_code << 22)             // DWORDs between pads
        | (pad_amount_code << 25);              // pad DWORDs - 1
  g1.s1 = (tile_w & 0xFFFFu) << 16;             // tensor_dim0[15:0] (==tile_w)
  g1.s2 = (tile_w >> 16) | ((tile_h & 0xFFFFu) << 16);  // dim0 hi | dim1 lo
  g1.s3 = (tile_h >> 16) | (tile_w << 16);      // dim1 hi | tile_dim0
  g1.s4 = tile_h;                               // tile_dim1 | tile_dim2=0
  g1.s5 = row_stride;                           // tensor_dim0_stride[31:0]
  g1.s6 = 0u;                                   // stride hi | dim1_stride lo
  g1.s7 = 0u;
  asm volatile("tensor_load_to_lds %0, %1" :: "s"(g0), "s"(g1) : "memory");
}

// ---------------------------------------------------------------------------
// WMMA GEMM: C[8192][5120] = A[8192][1024] * Wx[5120][1024]^T + bx
// Block: 256 threads (8 waves), tile 128x128, BK=32, TDM double-buffered.
// Wave grid 2(M) x 4(N): each wave owns 64x32 = 4x2 WMMA frags.
// ---------------------------------------------------------------------------
#define BM 128
#define BN 128
#define BK 32
#define LDT (BK + 8)   // 40 halfs = 80 B row stride (16B-aligned, padded)
#define NSTAGE (IN_DIM / BK)

__global__ __launch_bounds__(256)
void xproj_wmma_kernel(const unsigned short* __restrict__ A,
                       const unsigned short* __restrict__ B,
                       const float* __restrict__ bx,
                       float* __restrict__ C) {
  __shared__ __align__(16) unsigned short As[2][BM * LDT];  // 2 x 10240 B
  __shared__ __align__(16) unsigned short Bs[2][BN * LDT];

  const int tid  = threadIdx.x;
  const int lane = tid & 31;
  const int wave = tid >> 5;     // 0..7
  const int wm   = wave & 1;     // 64-row slab
  const int wn   = wave >> 1;    // 32-col slab
  const int r16  = lane & 15;
  const int hi   = lane >> 4;    // K-half selector per ISA 16-bit layout

  const int tileM = blockIdx.y * BM;
  const int tileN = blockIdx.x * BN;
  const unsigned short* Ag = A + (size_t)tileM * IN_DIM;
  const unsigned short* Bg = B + (size_t)tileN * IN_DIM;

  v8f acc[4][2];
#pragma unroll
  for (int tm = 0; tm < 4; ++tm)
#pragma unroll
    for (int tn = 0; tn < 2; ++tn)
      acc[tm][tn] = (v8f){0.f, 0.f, 0.f, 0.f, 0.f, 0.f, 0.f, 0.f};

  // Row = 32 elems * 2B = 64 B = 16 DWORDs -> pad_interval code 3 (2^4).
  // Pad = 16 B = 4 DWORDs -> pad_amount code 3.
  if (wave == 0) {
    tdm_load_tile_bf16(lds_off(&As[0][0]), Ag, BK, BM, IN_DIM, 3u, 3u);
    tdm_load_tile_bf16(lds_off(&Bs[0][0]), Bg, BK, BN, IN_DIM, 3u, 3u);
  }

  for (int s = 0; s < NSTAGE; ++s) {
    if (wave == 0) {
      if (s + 1 < NSTAGE) {
        int k0n = (s + 1) * BK;
        tdm_load_tile_bf16(lds_off(&As[(s + 1) & 1][0]), Ag + k0n,
                           BK, BM, IN_DIM, 3u, 3u);
        tdm_load_tile_bf16(lds_off(&Bs[(s + 1) & 1][0]), Bg + k0n,
                           BK, BN, IN_DIM, 3u, 3u);
        // TDM ops from one wave complete in order: <=2 outstanding means
        // the two stage-s tiles have fully landed in LDS.
        __builtin_amdgcn_s_wait_tensorcnt(2);
      } else {
        __builtin_amdgcn_s_wait_tensorcnt(0);
      }
    }
    __syncthreads();

    const unsigned short* Ac = &As[s & 1][0];
    const unsigned short* Bc = &Bs[s & 1][0];
    // ISA 16-bit operand layout: lanes 0-15 hold K=[0..7],[16..23];
    //                            lanes 16-31 hold K=[8..15],[24..31].
    v16bf a[4], b[2];
#pragma unroll
    for (int tm = 0; tm < 4; ++tm) {
      const unsigned short* p = &Ac[(wm * 64 + tm * 16 + r16) * LDT + hi * 8];
      union { v16bf v; v8bf h[2]; } u;
      u.h[0] = *(const v8bf*)p;
      u.h[1] = *(const v8bf*)(p + 16);
      a[tm] = u.v;
    }
#pragma unroll
    for (int tn = 0; tn < 2; ++tn) {
      const unsigned short* p = &Bc[(wn * 32 + tn * 16 + r16) * LDT + hi * 8];
      union { v16bf v; v8bf h[2]; } u;
      u.h[0] = *(const v8bf*)p;
      u.h[1] = *(const v8bf*)(p + 16);
      b[tn] = u.v;
    }
#pragma unroll
    for (int tm = 0; tm < 4; ++tm)
#pragma unroll
      for (int tn = 0; tn < 2; ++tn)
        acc[tm][tn] = __builtin_amdgcn_wmma_f32_16x16x32_bf16(
            false, a[tm], false, b[tn], (short)0, acc[tm][tn], false, false);
    __syncthreads();  // all readers done before wave 0 overwrites this buffer
  }

  // Epilogue per ISA C/D layout: VGPR i -> M = i + 8*hi, N = r16.
#pragma unroll
  for (int tm = 0; tm < 4; ++tm)
#pragma unroll
    for (int tn = 0; tn < 2; ++tn) {
      int row = tileM + wm * 64 + tm * 16 + hi * 8;
      int col = tileN + wn * 32 + tn * 16 + r16;
      float bias = bx[col];
#pragma unroll
      for (int i = 0; i < 8; ++i)
        C[(size_t)(row + i) * NGATE5 + col] = acc[tm][tn][i] + bias;
    }
}

// ---------------------------------------------------------------------------
// Persistent scan kernel: 64 blocks x 256 threads = 16384 threads.
// Phase A: hp = Wh@h + bh (4 thr/row), a = sigmoid(zx+zh)*tanh(c).
// Phase B: wmv = Wm@a + bm (16 thr/row), gates, state update.
// h and a (4 KB each) staged global->LDS with async loads each step.
// ---------------------------------------------------------------------------
#define SCAN_BLOCKS  64
#define SCAN_THREADS 256

__device__ __forceinline__ float sigmoid_(float x) {
  return 1.0f / (1.0f + __expf(-x));
}

__device__ __forceinline__ void grid_sync(unsigned* bar, unsigned* gen,
                                          unsigned nblk) {
  __syncthreads();
  if (threadIdx.x == 0) {
    __threadfence();
    volatile unsigned* vgen = gen;
    unsigned g = *vgen;
    unsigned prev = atomicAdd(bar, 1u);
    if (prev == nblk - 1u) {
      *bar = 0u;
      __threadfence();
      atomicAdd(gen, 1u);
    } else {
      while (*vgen == g) { __builtin_amdgcn_s_sleep(1); }
    }
    __threadfence();
  }
  __syncthreads();
}

// Async-copy 4 KB vector (1024 f32) global -> LDS: one b128 per thread.
__device__ __forceinline__ void stage_vec_async(float* lds_dst,
                                                const float* gsrc) {
  unsigned ldso = lds_off(lds_dst) + (unsigned)threadIdx.x * 16u;
  unsigned voff = (unsigned)threadIdx.x * 16u;
  asm volatile("global_load_async_to_lds_b128 %0, %1, %2"
               :: "v"(ldso), "v"(voff), "s"(gsrc) : "memory");
  asm volatile("s_wait_asynccnt 0x0" ::: "memory");
  __syncthreads();
}

__global__ __launch_bounds__(SCAN_THREADS)
void ulstm_scan_kernel(const float* __restrict__ xproj,  // [T][5120]
                       const float* __restrict__ Wh,     // [4096][1024]
                       const float* __restrict__ bh,
                       const float* __restrict__ Wm,     // [1024][1024]
                       const float* __restrict__ bm,
                       float* __restrict__ hpbuf,        // 4096
                       float* __restrict__ abuf,         // 1024
                       float* __restrict__ cbuf,         // 1024
                       float* __restrict__ hbuf,         // 1024
                       unsigned* __restrict__ bar,
                       unsigned* __restrict__ gen,
                       float* __restrict__ out) {
  __shared__ __align__(16) float hls[MEM_DIM];
  __shared__ __align__(16) float als[MEM_DIM];

  const int gtid  = blockIdx.x * SCAN_THREADS + threadIdx.x;  // 0..16383
  const int rowA  = gtid >> 2, partA = gtid & 3;
  const int rowB  = gtid >> 4, partB = gtid & 15;

  for (int t = 0; t < T_STEPS; ++t) {
    const float* xrow = xproj + (size_t)t * NGATE5;

    // ---- Phase A: hp = Wh @ h + bh ; a = sigmoid(zx+zh)*tanh(c) ----
    stage_vec_async(hls, hbuf);
    {
      const float4* w4 = (const float4*)(Wh + (size_t)rowA * MEM_DIM) + partA * 64;
      const float4* h4 = (const float4*)hls + partA * 64;
      float s = 0.f;
#pragma unroll 8
      for (int k = 0; k < 64; ++k) {
        float4 w = w4[k], h = h4[k];
        s += w.x * h.x + w.y * h.y + w.z * h.z + w.w * h.w;
      }
      s += __shfl_xor(s, 1, 32);
      s += __shfl_xor(s, 2, 32);
      if (partA == 0) {
        float hp = s + bh[rowA];
        hpbuf[rowA] = hp;
        int j = rowA - 2 * MEM_DIM;
        if (j >= 0 && j < MEM_DIM) {
          float z = sigmoid_(xrow[2 * MEM_DIM + j] + hp);
          abuf[j] = z * tanhf(cbuf[j]);
        }
      }
    }
    grid_sync(bar, gen, SCAN_BLOCKS);

    // ---- Phase B: wmv = Wm @ a + bm ; gates ; state update ----
    stage_vec_async(als, abuf);
    {
      const float4* w4 = (const float4*)(Wm + (size_t)rowB * MEM_DIM) + partB * 16;
      const float4* a4 = (const float4*)als + partB * 16;
      float s = 0.f;
#pragma unroll 8
      for (int k = 0; k < 16; ++k) {
        float4 w = w4[k], a = a4[k];
        s += w.x * a.x + w.y * a.y + w.z * a.z + w.w * a.w;
      }
      s += __shfl_xor(s, 1, 32);
      s += __shfl_xor(s, 2, 32);
      s += __shfl_xor(s, 4, 32);
      s += __shfl_xor(s, 8, 32);
      if (partB == 0) {
        int j = rowB;
        float wmv = s + bm[j];
        float gi = sigmoid_(xrow[j] + hpbuf[j]);
        float go = sigmoid_(xrow[MEM_DIM + j] + hpbuf[MEM_DIM + j]);
        float gf = sigmoid_(xrow[3 * MEM_DIM + j] + hpbuf[3 * MEM_DIM + j]);
        float gu = tanhf(xrow[4 * MEM_DIM + j] + wmv);
        float c2 = gi * gu + gf * cbuf[j];
        cbuf[j] = c2;
        hbuf[j] = go * tanhf(c2);
      }
    }
    grid_sync(bar, gen, SCAN_BLOCKS);
  }

  if (gtid < MEM_DIM) {
    out[gtid] = cbuf[gtid];
    out[MEM_DIM + gtid] = hbuf[gtid];
  }
}

// ---------------------------------------------------------------------------
__global__ void scan_init_kernel(float* cbuf, float* hbuf,
                                 unsigned* bar, unsigned* gen) {
  int i = threadIdx.x;  // 1024 threads
  cbuf[i] = 0.f;
  hbuf[i] = 0.f;
  if (i == 0) { *bar = 0u; *gen = 0u; }
}

// ---------------------------------------------------------------------------
extern "C" void kernel_launch(void* const* d_in, const int* in_sizes, int n_in,
                              void* d_out, int out_size, void* d_ws,
                              size_t ws_size, hipStream_t stream) {
  (void)in_sizes; (void)n_in; (void)out_size; (void)ws_size;
  const float* inputs = (const float*)d_in[0];
  const float* Wx     = (const float*)d_in[1];
  const float* bx     = (const float*)d_in[2];
  const float* Wh     = (const float*)d_in[3];
  const float* bh     = (const float*)d_in[4];
  const float* Wm     = (const float*)d_in[5];
  const float* bm     = (const float*)d_in[6];
  float* out = (float*)d_out;

  // Workspace layout (~195.3 MB total)
  char* p = (char*)d_ws;
  float* xproj = (float*)p;             p += (size_t)T_STEPS * NGATE5 * sizeof(float);
  unsigned short* inb = (unsigned short*)p; p += (size_t)T_STEPS * IN_DIM * 2;
  unsigned short* wxb = (unsigned short*)p; p += (size_t)NGATE5 * IN_DIM * 2;
  float* hpbuf = (float*)p;             p += NGATE4 * sizeof(float);
  float* abuf  = (float*)p;             p += MEM_DIM * sizeof(float);
  float* cbuf  = (float*)p;             p += MEM_DIM * sizeof(float);
  float* hbuf  = (float*)p;             p += MEM_DIM * sizeof(float);
  unsigned* bar = (unsigned*)p;
  unsigned* gen = (unsigned*)(p + 64);  // separate cachelines

  scan_init_kernel<<<1, 1024, 0, stream>>>(cbuf, hbuf, bar, gen);

  f32_to_bf16_kernel<<<4096, 256, 0, stream>>>(inputs, inb, T_STEPS * IN_DIM);
  f32_to_bf16_kernel<<<4096, 256, 0, stream>>>(Wx, wxb, NGATE5 * IN_DIM);

  dim3 grid(NGATE5 / BN, T_STEPS / BM);  // (40, 64)
  xproj_wmma_kernel<<<grid, 256, 0, stream>>>(inb, wxb, bx, xproj);

  ulstm_scan_kernel<<<SCAN_BLOCKS, SCAN_THREADS, 0, stream>>>(
      xproj, Wh, bh, Wm, bm, hpbuf, abuf, cbuf, hbuf, bar, gen, out);
}
